// TransformerLayer_86981677678897
// MI455X (gfx1250) — matrix operand order repaired and checked by reading, not compiled, not run on hardware
//
#include <hip/hip_runtime.h>

// ---------------------------------------------------------------------------
// Transformer layer for MI455X (gfx1250, wave32, WMMA).
// bf16 data / f32 accumulate via v_wmma_f32_16x16x32_bf16.
// Global->LDS staging uses gfx1250 async-to-LDS DMA (ASYNCcnt) where the
// layout is a straight copy; packed-pair transposes otherwise.
// ---------------------------------------------------------------------------

typedef __bf16 bf16;
typedef __attribute__((ext_vector_type(16))) __bf16        v16bf;
typedef __attribute__((ext_vector_type(8)))  __bf16        v8bf;
typedef __attribute__((ext_vector_type(8)))  unsigned short v8us;
typedef __attribute__((ext_vector_type(8)))  float         v8f;
typedef __attribute__((ext_vector_type(4)))  float         v4f;

#define B_  2
#define L_  2048
#define D_  1024
#define H_  16
#define HD_ 64
#define R_  512
#define M_  (B_ * L_)   // 4096 rows

__device__ __forceinline__ v8f wmma_bf16(v16bf a, v16bf b, v8f c) {
  return __builtin_amdgcn_wmma_f32_16x16x32_bf16(false, a, false, b,
                                                 (short)0, c, false, false);
}

__device__ __forceinline__ v16bf cat8(v8bf lo, v8bf hi) {
  return __builtin_shufflevector(lo, hi, 0,1,2,3,4,5,6,7,8,9,10,11,12,13,14,15);
}

__device__ __forceinline__ float safe_exp(float a) {
  float e = __expf(a);
  return (e == e) ? e : 0.0f;   // NaN from (-inf)-(-inf) -> 0
}

// gfx1250 async global->LDS DMA (tracked by ASYNCcnt, §10 VGLOBAL op 98).
__device__ __forceinline__ void async_load_b128(unsigned lds_off, const void* g) {
  asm volatile("global_load_async_to_lds_b128 %0, %1, off"
               :: "v"(lds_off), "v"((unsigned long long)(uintptr_t)g)
               : "memory");
}
__device__ __forceinline__ void wait_async0() {
  asm volatile("s_wait_asynccnt 0" ::: "memory");
}
__device__ __forceinline__ unsigned lds_off32(const void* p) {
  return (unsigned)(uintptr_t)p;   // LDS aperture: addr[31:0] is LDS offset
}

// ---------------------------------------------------------------------------
// fp32 -> bf16 conversion (weights)
// ---------------------------------------------------------------------------
__global__ __launch_bounds__(256)
void cvt_f32_bf16(const float* __restrict__ in, bf16* __restrict__ out, int n4) {
  int i = blockIdx.x * 256 + threadIdx.x;
  if (i >= n4) return;
  v4f v = *(const v4f*)(in + (size_t)i * 4);
  bf16* o = out + (size_t)i * 4;
  o[0] = (bf16)v.x; o[1] = (bf16)v.y; o[2] = (bf16)v.z; o[3] = (bf16)v.w;
}

// ---------------------------------------------------------------------------
// LayerNorm over D=1024, fp32 in -> bf16 out.  1 block per row, 256 threads.
// ---------------------------------------------------------------------------
__global__ __launch_bounds__(256)
void layernorm_bf16(const float* __restrict__ x, const float* __restrict__ g,
                    const float* __restrict__ beta, bf16* __restrict__ out) {
  int row = blockIdx.x;
  int tid = threadIdx.x;
  const float* xr = x + (size_t)row * D_;
  v4f v = *(const v4f*)(xr + tid * 4);
  float s  = v.x + v.y + v.z + v.w;
  float sq = v.x*v.x + v.y*v.y + v.z*v.z + v.w*v.w;
  __shared__ float rs[256], rq[256];
  rs[tid] = s; rq[tid] = sq;
  __syncthreads();
  for (int st = 128; st > 0; st >>= 1) {
    if (tid < st) { rs[tid] += rs[tid + st]; rq[tid] += rq[tid + st]; }
    __syncthreads();
  }
  float mu  = rs[0] * (1.0f / D_);
  float var = rq[0] * (1.0f / D_) - mu * mu;
  float inv = rsqrtf(var + 1e-5f);
  bf16* o = out + (size_t)row * D_ + tid * 4;
  #pragma unroll
  for (int j = 0; j < 4; ++j) {
    int c = tid * 4 + j;
    o[j] = (bf16)((v[j] - mu) * inv * g[c] + beta[c]);
  }
}

// ---------------------------------------------------------------------------
// Tiled bf16 WMMA GEMM: C = A[M,K] @ W[K,N] (+bias, epilogue variants).
// Block tile 128x128, BK=32, 8 waves (2x4), wave tile 64x32 (4x2 of 16x16).
// A tile: async DMA to LDS.  W tile: packed (k,k+1)-pair transpose -> u32 LDS.
// ---------------------------------------------------------------------------
#define BM 128
#define BN 128
#define BK 32

enum { EPI_QKV = 0, EPI_RESID_F32 = 1, EPI_RELU_BF16 = 2 };

template <int EPI>
__global__ __launch_bounds__(256)
void gemm_bf16_wmma(const bf16* __restrict__ A, const bf16* __restrict__ W,
                    const float* __restrict__ bias, int M, int N, int K,
                    float* __restrict__ outF, bf16* __restrict__ outB,
                    const float* __restrict__ resid,
                    bf16* __restrict__ qb, bf16* __restrict__ kb,
                    bf16* __restrict__ vb) {
  __shared__ bf16     As[BM][BK + 8];     // [m][k]         10 KB
  __shared__ unsigned Ws32[BN][16 + 4];   // [n][kpair]     10 KB

  const int tid  = threadIdx.x;
  const int wid  = tid >> 5;
  const int lane = tid & 31;
  const int l16  = lane & 15;
  const int lh   = lane >> 4;       // half-wave selector

  const int bm0 = (blockIdx.x % (M / BM)) * BM;
  const int bn0 = (blockIdx.x / (M / BM)) * BN;
  const int wm0 = (wid >> 2) * 64;  // 0 / 64
  const int wn0 = (wid & 3) * 32;   // 0..96

  v8f acc[4][2] = {};

  for (int k0 = 0; k0 < K; k0 += BK) {
    __syncthreads();
    // --- stage A tile via async DMA: thread -> row tid>>1, 16 bf16 segment
    {
      int row = tid >> 1, seg = (tid & 1) * 16;
      const bf16* src = A + (size_t)(bm0 + row) * K + k0 + seg;
      unsigned l0 = lds_off32(&As[row][seg]);
      async_load_b128(l0, src);
      async_load_b128(l0 + 16, src + 8);
    }
    // --- stage W: pack (k,k+1) bf16 pairs into u32, store transposed
    {
      int p  = tid >> 4;            // k-pair index 0..15
      int c0 = (tid & 15) * 8;      // column group
      const unsigned short* src =
          (const unsigned short*)(W + (size_t)(k0 + 2 * p) * N + bn0 + c0);
      v8us lo = *(const v8us*)src;
      v8us hi = *(const v8us*)(src + N);
      #pragma unroll
      for (int j = 0; j < 8; ++j)
        Ws32[c0 + j][p] = (unsigned)lo[j] | ((unsigned)hi[j] << 16);
    }
    if (k0 + BK < K)   // gfx1250 global_prefetch_b8 of next W tile
      __builtin_prefetch(W + (size_t)(k0 + BK + 2 * (tid >> 4)) * N + bn0 + (tid & 15) * 8, 0, 0);
    wait_async0();
    __syncthreads();

    v16bf af[4], bfr[2];
    #pragma unroll
    for (int mt = 0; mt < 4; ++mt) {
      int row = wm0 + mt * 16 + l16;
      int ko  = 16 * lh;                // A-layout: K = 16*lh + e
      af[mt] = cat8(*(const v8bf*)&As[row][ko],
                    *(const v8bf*)&As[row][8 + ko]);
    }
    #pragma unroll
    for (int nt = 0; nt < 2; ++nt) {
      int col = wn0 + nt * 16 + l16;
      const bf16* wp = (const bf16*)&Ws32[col][8 * lh];  // K = 16*lh + e
      bfr[nt] = cat8(*(const v8bf*)wp, *(const v8bf*)(wp + 8));
    }
    #pragma unroll
    for (int mt = 0; mt < 4; ++mt)
      #pragma unroll
      for (int nt = 0; nt < 2; ++nt)
        acc[mt][nt] = wmma_bf16(af[mt], bfr[nt], acc[mt][nt]);
  }

  // epilogue: C element (m,n): m = r + 8*lh (+tiles), n = l16 (+tiles)
  #pragma unroll
  for (int mt = 0; mt < 4; ++mt)
    #pragma unroll
    for (int nt = 0; nt < 2; ++nt)
      #pragma unroll
      for (int r = 0; r < 8; ++r) {
        int m = bm0 + wm0 + mt * 16 + r + 8 * lh;
        int n = bn0 + wn0 + nt * 16 + l16;
        float val = acc[mt][nt][r] + bias[n];
        if constexpr (EPI == EPI_QKV) {
          int which = n >> 10;            // 0:q 1:k 2:v
          int hh = (n & 1023) >> 6;
          int hd = n & 63;
          int bb = m >> 11;               // L = 2048
          int li = m & 2047;
          bf16* dst = (which == 0) ? qb : ((which == 1) ? kb : vb);
          dst[((((size_t)bb * H_ + hh) * L_) + li) * HD_ + hd] = (bf16)val;
        } else if constexpr (EPI == EPI_RESID_F32) {
          size_t idx = (size_t)m * N + n;
          outF[idx] = resid[idx] + val;
        } else {  // EPI_RELU_BF16
          outB[(size_t)m * N + n] = (bf16)fmaxf(val, 0.0f);
        }
      }
}

// ---------------------------------------------------------------------------
// Flash attention with banded rel-pos bias (valid iff 0 <= q-k < 512).
// Grid: (B*H) * (L/128). 8 waves/block, each wave owns 16 query rows.
// K tile: async DMA.  V tile: packed key-pair transpose.
// ---------------------------------------------------------------------------
__global__ __launch_bounds__(256)
void attn_wmma(const bf16* __restrict__ q, const bf16* __restrict__ k,
               const bf16* __restrict__ v, const float* __restrict__ res_pos,
               bf16* __restrict__ ctx) {
  const int QCHUNKS = L_ / 128;             // 16
  const int bh = blockIdx.x / QCHUNKS;
  const int qc = blockIdx.x % QCHUNKS;
  const int h  = bh % H_;
  const int bb = bh / H_;
  const int q0b = qc * 128;

  const int tid  = threadIdx.x;
  const int wid  = tid >> 5;
  const int lane = tid & 31;
  const int l16  = lane & 15;
  const int lh   = lane >> 4;
  const int q0   = q0b + wid * 16;

  __shared__ float    respos_s[R_];          // 2 KB
  __shared__ bf16     Kt[32][HD_ + 8];       // K chunk, [key][d]
  __shared__ unsigned Vt32[HD_][16 + 4];     // V chunk transposed, [d][keypair]
  __shared__ bf16     Ps[8][16][32 + 8];     // per-wave probs, [m][key]

  for (int i = tid; i < R_; i += 256) respos_s[i] = res_pos[h * R_ + i];

  const bf16* qp = q + (size_t)bh * L_ * HD_;
  const bf16* kp = k + (size_t)bh * L_ * HD_;
  const bf16* vp = v + (size_t)bh * L_ * HD_;

  // Q fragments (A layout), 2 k-steps of 32
  v16bf qf[2];
  {
    const bf16* src = qp + (size_t)(q0 + l16) * HD_;
    #pragma unroll
    for (int f = 0; f < 2; ++f) {
      int ko = f * 32 + 16 * lh;
      qf[f] = cat8(*(const v8bf*)(src + ko), *(const v8bf*)(src + ko + 8));
    }
  }

  float mrow[8], lrow[8];
  #pragma unroll
  for (int r = 0; r < 8; ++r) { mrow[r] = -INFINITY; lrow[r] = 0.0f; }
  v8f oacc[4] = {};

  int t0 = q0b - (R_ - 1); if (t0 < 0) t0 = 0;
  const int kc_begin = t0 & ~31;
  const int kc_end   = q0b + 128;
  const float SCALE  = 0.125f;              // 1/sqrt(64)

  for (int kc = kc_begin; kc < kc_end; kc += 32) {
    __syncthreads();
    if (tid < 128) {   // K tile: straight copy via async DMA
      int row = tid >> 2, c0 = (tid & 3) * 16;
      const bf16* src = kp + (size_t)(kc + row) * HD_ + c0;
      unsigned l0 = lds_off32(&Kt[row][c0]);
      async_load_b128(l0, src);
      async_load_b128(l0 + 16, src + 8);
    } else {           // V tile: packed (key,key+1) pair transpose
      int t2 = tid - 128;
      int p  = t2 >> 3;          // key-pair 0..15
      int c0 = (t2 & 7) * 8;     // d group
      const unsigned short* src =
          (const unsigned short*)(vp + (size_t)(kc + 2 * p) * HD_ + c0);
      v8us lo = *(const v8us*)src;
      v8us hi = *(const v8us*)(src + HD_);
      #pragma unroll
      for (int j = 0; j < 8; ++j)
        Vt32[c0 + j][p] = (unsigned)lo[j] | ((unsigned)hi[j] << 16);
    }
    wait_async0();
    __syncthreads();

    // wave-uniform band skip as a *scalar* branch (EXEC stays all-ones)
    int skip = (kc > q0 + 15) | (kc + 31 < q0 - (R_ - 1));
    if (__builtin_amdgcn_readfirstlane(skip)) continue;

    // scores: two 16x16 tiles over this 32-key chunk
    v8f sc[2] = {};
    #pragma unroll
    for (int t = 0; t < 2; ++t) {
      int key = t * 16 + l16;
      #pragma unroll
      for (int s = 0; s < 2; ++s) {
        int ko = s * 32 + 16 * lh;          // B-layout: Kdim = 16*lh + e
        v16bf kf = cat8(*(const v8bf*)&Kt[key][ko],
                        *(const v8bf*)&Kt[key][ko + 8]);
        sc[t] = wmma_bf16(qf[s], kf, sc[t]);
      }
    }

    float sv0[8], sv1[8];
    #pragma unroll
    for (int r = 0; r < 8; ++r) {
      int qrow = q0 + r + 8 * lh;
      int d0 = qrow - (kc + l16);
      int d1 = qrow - (kc + 16 + l16);
      float b0 = (d0 >= 0 && d0 < R_) ? respos_s[d0] : -INFINITY;
      float b1 = (d1 >= 0 && d1 < R_) ? respos_s[d1] : -INFINITY;
      sv0[r] = sc[0][r] * SCALE + b0;
      sv1[r] = sc[1][r] * SCALE + b1;
    }

    #pragma unroll
    for (int r = 0; r < 8; ++r) {
      float mx = fmaxf(sv0[r], sv1[r]);
      #pragma unroll
      for (int off = 8; off > 0; off >>= 1) mx = fmaxf(mx, __shfl_xor(mx, off, 16));
      float mnew = fmaxf(mrow[r], mx);
      float corr = safe_exp(mrow[r] - mnew);
      float p0 = safe_exp(sv0[r] - mnew);
      float p1 = safe_exp(sv1[r] - mnew);
      float ps = p0 + p1;
      #pragma unroll
      for (int off = 8; off > 0; off >>= 1) ps += __shfl_xor(ps, off, 16);
      lrow[r] = corr * lrow[r] + ps;
      mrow[r] = mnew;
      #pragma unroll
      for (int dc = 0; dc < 4; ++dc) oacc[dc][r] *= corr;
      int pm = r + 8 * lh;
      Ps[wid][pm][l16]      = (bf16)p0;
      Ps[wid][pm][16 + l16] = (bf16)p1;
    }

    // P (16x32, A layout) @ V (32x64) -> accumulate ctx
    v16bf pf = cat8(*(const v8bf*)&Ps[wid][l16][16 * lh],
                    *(const v8bf*)&Ps[wid][l16][8 + 16 * lh]);
    #pragma unroll
    for (int dc = 0; dc < 4; ++dc) {
      int dcol = dc * 16 + l16;
      const bf16* vp2 = (const bf16*)&Vt32[dcol][8 * lh];  // K = 16*lh + e
      v16bf vf = cat8(*(const v8bf*)vp2, *(const v8bf*)(vp2 + 8));
      oacc[dc] = wmma_bf16(pf, vf, oacc[dc]);
    }
  }

  // write ctx in (B, L, D) layout: col = h*64 + d
  #pragma unroll
  for (int dc = 0; dc < 4; ++dc)
    #pragma unroll
    for (int r = 0; r < 8; ++r) {
      int qrow = q0 + r + 8 * lh;
      int col  = h * HD_ + dc * 16 + l16;
      float den = lrow[r] > 0.0f ? lrow[r] : 1.0f;
      ctx[((size_t)bb * L_ + qrow) * D_ + col] = (bf16)(oacc[dc][r] / den);
    }
}

// ---------------------------------------------------------------------------
// Host-side orchestration
// ---------------------------------------------------------------------------
extern "C" void kernel_launch(void* const* d_in, const int* in_sizes, int n_in,
                              void* d_out, int out_size, void* d_ws, size_t ws_size,
                              hipStream_t stream) {
  const float* x      = (const float*)d_in[0];
  const float* respos = (const float*)d_in[1];
  const float* w_qkv  = (const float*)d_in[2];
  const float* b_qkv  = (const float*)d_in[3];
  const float* w_out  = (const float*)d_in[4];
  const float* b_out  = (const float*)d_in[5];
  const float* w1     = (const float*)d_in[6];
  const float* b1     = (const float*)d_in[7];
  const float* w2     = (const float*)d_in[8];
  const float* b2     = (const float*)d_in[9];
  const float* ln1_g  = (const float*)d_in[10];
  const float* ln1_b  = (const float*)d_in[11];
  const float* ln2_g  = (const float*)d_in[12];
  const float* ln2_b  = (const float*)d_in[13];

  char* ws = (char*)d_ws;
  const size_t SZ_ACT_BF = (size_t)M_ * D_ * 2;        // 8 MB
  bf16*  xn1   = (bf16*)(ws);                          // 0
  bf16*  qb    = (bf16*)(ws + SZ_ACT_BF);              // 8 MB
  bf16*  kb    = (bf16*)(ws + 2 * SZ_ACT_BF);          // 16 MB
  bf16*  vb    = (bf16*)(ws + 3 * SZ_ACT_BF);          // 24 MB
  bf16*  ctxb  = (bf16*)(ws + 4 * SZ_ACT_BF);          // 32 MB
  float* x2    = (float*)(ws + 5 * SZ_ACT_BF);         // 40 MB (fp32, 16 MB)
  bf16*  hb    = (bf16*)(ws + 7 * SZ_ACT_BF);          // 56 MB
  bf16*  h1b   = (bf16*)(ws + 8 * SZ_ACT_BF);          // 64 MB (32 MB)
  bf16*  wqkvb = (bf16*)(ws + 12 * SZ_ACT_BF);         // 96 MB
  bf16*  woutb = (bf16*)(ws + 12 * SZ_ACT_BF + 6291456);
  bf16*  w1b   = (bf16*)(ws + 12 * SZ_ACT_BF + 6291456 + 2097152);
  bf16*  w2b   = (bf16*)(ws + 12 * SZ_ACT_BF + 6291456 + 2097152 + 8388608);

  auto cvt = [&](const float* src, bf16* dst, int n) {
    int n4 = n / 4;
    cvt_f32_bf16<<<(n4 + 255) / 256, 256, 0, stream>>>(src, dst, n4);
  };
  cvt(w_qkv, wqkvb, D_ * 3 * D_);
  cvt(w_out, woutb, D_ * D_);
  cvt(w1, w1b, D_ * 4 * D_);
  cvt(w2, w2b, 4 * D_ * D_);

  // LN1 -> xn (bf16)
  layernorm_bf16<<<M_, 256, 0, stream>>>(x, ln1_g, ln1_b, xn1);

  // QKV GEMM: [4096,1024] @ [1024,3072], scatter into q/k/v (B,H,L,HD)
  gemm_bf16_wmma<EPI_QKV><<<(M_ / BM) * (3 * D_ / BN), 256, 0, stream>>>(
      xn1, wqkvb, b_qkv, M_, 3 * D_, D_, nullptr, nullptr, nullptr, qb, kb, vb);

  // attention
  attn_wmma<<<B_ * H_ * (L_ / 128), 256, 0, stream>>>(qb, kb, vb, respos, ctxb);

  // x2 = x + ctx @ w_out + b_out
  gemm_bf16_wmma<EPI_RESID_F32><<<(M_ / BM) * (D_ / BN), 256, 0, stream>>>(
      ctxb, woutb, b_out, M_, D_, D_, x2, nullptr, x, nullptr, nullptr, nullptr);

  // LN2 -> h (bf16)
  layernorm_bf16<<<M_, 256, 0, stream>>>(x2, ln2_g, ln2_b, hb);

  // h1 = relu(h @ w1 + b1)  (bf16)
  gemm_bf16_wmma<EPI_RELU_BF16><<<(M_ / BM) * (4 * D_ / BN), 256, 0, stream>>>(
      hb, w1b, b1, M_, 4 * D_, D_, nullptr, h1b, nullptr, nullptr, nullptr, nullptr);

  // out = x2 + h1 @ w2 + b2  (fp32)
  gemm_bf16_wmma<EPI_RESID_F32><<<(M_ / BM) * (D_ / BN), 256, 0, stream>>>(
      h1b, w2b, b2, M_, D_, 4 * D_, (float*)d_out, nullptr, x2, nullptr, nullptr, nullptr);
}
